// MambaLayer_3702261809512
// MI455X (gfx1250) — compile-verified, matrix operand order checked by reading, not run on hardware
//
#include <hip/hip_runtime.h>
#include <hip/hip_bf16.h>
#include <math.h>

// ---------------- model constants ----------------
constexpr int BATCH = 2;
constexpr int LSEQ  = 1024;
constexpr int DM    = 512;              // d_model
constexpr int DI    = 1024;             // d_inner
constexpr int DSTATE= 16;
constexpr int DTR   = 32;               // dt_rank
constexpr int NSL   = 32;               // n slices
constexpr int LC    = LSEQ / NSL;       // 32
constexpr int DCONV = 4;
constexpr int TOK   = BATCH * LSEQ;     // 2048 tokens

typedef __attribute__((ext_vector_type(16))) __bf16 v16bf;
typedef __attribute__((ext_vector_type(8)))  float  v8f;

// round-to-nearest-even bf16 sitting in the HIGH half of the dword
__device__ __forceinline__ unsigned bfrne(float f) {
    unsigned u = __builtin_bit_cast(unsigned, f);
    return u + 0x7fffu + ((u >> 16) & 1u);
}
// pack two bf16 into one dword with a single v_perm_b32 (a -> low, b -> high)
__device__ __forceinline__ unsigned pack2bf(float a, float b) {
    return __builtin_amdgcn_perm(bfrne(b), bfrne(a), 0x07060302u);
}

__device__ __forceinline__ float siluf(float x) {
    return x / (1.0f + __expf(-x));
}

// branch position -> original sequence index
//  0: forward (identity)  1: backward (flip)  2: sliced (32x32 transpose)
__device__ __forceinline__ int map_pos(int branch, int p) {
    if (branch == 0) return p;
    if (branch == 1) return LSEQ - 1 - p;
    return (p & (NSL - 1)) * LC + (p >> 5);   // involution
}

// -----------------------------------------------------------------------
// Generic WMMA GEMM:  C[M,N] = act( A[M,K] * Bw[N,K]^T  (+ bias) )
// A row-major lda, Bw row-major ldb (weights are stored (N,K) in reference).
// Macro tile 128x64, 8 waves (4 M x 2 N), each wave register-blocks 2x2
// 16x16 tiles -> 4 v_wmma per 32-wide K-step. Staging: 128-bit global
// loads -> v_perm-packed bf16 pairs -> 64-bit DS stores.
// mode: 0 = plain store, 1 = softplus(x + bias[col])
// -----------------------------------------------------------------------
__global__ __launch_bounds__(256) void gemm_wmma_bf16(
    const float* __restrict__ A,  int lda,
    const float* __restrict__ Bw, int ldb,
    float* __restrict__ C,        int ldc,
    int K,
    const float* __restrict__ bias, int mode)
{
    __shared__ unsigned sAu[128 * 16];   // 128 rows x 32 bf16 (packed pairs)
    __shared__ unsigned sBu[64 * 16];    //  64 rows x 32 bf16 (packed pairs)

    const int m0   = blockIdx.x * 128;
    const int n0   = blockIdx.y * 64;
    const int tid  = threadIdx.x;
    const int wave = tid >> 5;
    const int lane = tid & 31;
    const int wm   = wave & 3;     // 32-row slice of the 128-row tile
    const int wn   = wave >> 2;    // 32-col slice of the 64-col tile

    v8f acc[2][2] = {};

    for (int k0 = 0; k0 < K; k0 += 32) {
        // ---- stage A tile 128x32: 1024 float4 loads, 4 per thread ----
#pragma unroll
        for (int j = 0; j < 4; ++j) {
            int i  = tid + 256 * j;          // 0..1023
            int r  = i >> 3;                 // row 0..127
            int kq = i & 7;                  // float4 index 0..7
            float4 v = *(const float4*)&A[(size_t)(m0 + r) * lda + k0 + kq * 4];
            ((uint2*)sAu)[r * 8 + kq] =
                make_uint2(pack2bf(v.x, v.y), pack2bf(v.z, v.w));
        }
        // ---- stage B tile 64x32: 512 float4 loads, 2 per thread ----
#pragma unroll
        for (int j = 0; j < 2; ++j) {
            int i  = tid + 256 * j;          // 0..511
            int r  = i >> 3;                 // row 0..63
            int kq = i & 7;
            float4 v = *(const float4*)&Bw[(size_t)(n0 + r) * ldb + k0 + kq * 4];
            ((uint2*)sBu)[r * 8 + kq] =
                make_uint2(pack2bf(v.x, v.y), pack2bf(v.z, v.w));
        }
        // prefetch next A K-slab into L2 (global_prefetch_b8 path)
        if (k0 + 32 < K)
            __builtin_prefetch(&A[(size_t)(m0 + (tid >> 1)) * lda + k0 + 32], 0, 1);
        __syncthreads();

        // ---- gather fragments per documented 16-bit 16x32 A layout ----
        // lanes 0-15: row = lane, K-halves {0..7,16..23}; lanes 16-31: +8
        const __bf16* sA = reinterpret_cast<const __bf16*>(sAu);
        const __bf16* sB = reinterpret_cast<const __bf16*>(sBu);
        const int lrow = lane & 15;
        const int lh   = (lane >> 4) << 3;

        v16bf af[2], bfv[2];
#pragma unroll
        for (int mi = 0; mi < 2; ++mi)
#pragma unroll
            for (int e = 0; e < 16; ++e) {
                int v = e >> 1, jj = e & 1;
                int k = (v < 4) ? (lh + 2 * v + jj) : (16 + lh + 2 * (v - 4) + jj);
                af[mi][e] = sA[(wm * 32 + mi * 16 + lrow) * 32 + k];
            }
#pragma unroll
        for (int ni = 0; ni < 2; ++ni)
#pragma unroll
            for (int e = 0; e < 16; ++e) {
                int v = e >> 1, jj = e & 1;
                int k = (v < 4) ? (lh + 2 * v + jj) : (16 + lh + 2 * (v - 4) + jj);
                bfv[ni][e] = sB[(wn * 32 + ni * 16 + lrow) * 32 + k];
            }
#pragma unroll
        for (int mi = 0; mi < 2; ++mi)
#pragma unroll
            for (int ni = 0; ni < 2; ++ni)
                acc[mi][ni] = __builtin_amdgcn_wmma_f32_16x16x32_bf16(
                    false, af[mi], false, bfv[ni], (short)0, acc[mi][ni],
                    false, false);
        __syncthreads();
    }

    // C/D layout: VGPR r -> M = r + (lane>=16)*8, N = lane&15
#pragma unroll
    for (int mi = 0; mi < 2; ++mi) {
#pragma unroll
        for (int ni = 0; ni < 2; ++ni) {
            const int col   = n0 + wn * 32 + ni * 16 + (lane & 15);
            const int rbase = m0 + wm * 32 + mi * 16 + ((lane >> 4) << 3);
#pragma unroll
            for (int r = 0; r < 8; ++r) {
                float v = acc[mi][ni][r];
                if (mode == 1) {
                    v += bias[col];
                    v = (v > 20.0f) ? v : log1pf(__expf(v));   // softplus
                }
                C[(size_t)(rbase + r) * ldc + col] = v;
            }
        }
    }
}

// -----------------------------------------------------------------------
// Causal depthwise conv (4 taps) + SiLU, reading xz (token-major, 2*DI wide)
// at branch-reordered positions; writes u (branch order, TOK x DI).
// -----------------------------------------------------------------------
__global__ __launch_bounds__(256) void conv_silu_kernel(
    const float* __restrict__ xz, const float* __restrict__ w,
    const float* __restrict__ bias, float* __restrict__ u, int branch)
{
    int idx = blockIdx.x * 256 + threadIdx.x;     // over TOK*DI
    int d   = idx & (DI - 1);
    int row = idx >> 10;                          // b*LSEQ + p (branch order)
    int b   = row >> 10;
    int p   = row & (LSEQ - 1);

    float acc = bias[d];
#pragma unroll
    for (int t = 0; t < DCONV; ++t) {
        int q = p - (DCONV - 1) + t;
        if (q >= 0) {
            int l = map_pos(branch, q);
            acc += w[d * DCONV + t] *
                   xz[(size_t)(b * LSEQ + l) * (2 * DI) + d];
        }
    }
    u[(size_t)row * DI + d] = siluf(acc);
}

// -----------------------------------------------------------------------
// Selective scan: one channel d per lane, 16 fp32 states in registers.
// B/C vectors (shared across d) staged in LDS in 16-step chunks.
// Gated by silu(z) and scattered back to the ORIGINAL sequence position.
// -----------------------------------------------------------------------
__global__ __launch_bounds__(256) void scan_kernel(
    const float* __restrict__ dt,   const float* __restrict__ xdbl,
    const float* __restrict__ u,    const float* __restrict__ xz,
    const float* __restrict__ Alog, const float* __restrict__ Dp,
    float* __restrict__ ysum, int branch, int accumulate)
{
    __shared__ float sB[16 * 16];
    __shared__ float sC[16 * 16];

    const int b = blockIdx.x >> 2;                       // batch
    const int d = (blockIdx.x & 3) * 256 + threadIdx.x;  // channel

    float Ac[16], h[16];
#pragma unroll
    for (int n = 0; n < 16; ++n) {
        Ac[n] = -__expf(Alog[d * 16 + n]);               // A = -exp(A_log)
        h[n]  = 0.0f;
    }
    const float Dv = Dp[d];

    for (int p0 = 0; p0 < LSEQ; p0 += 16) {
        __syncthreads();
        {   // stage 16 steps of B/C (16 n each): 256 threads, 1 B + 1 C each
            int i  = threadIdx.x;
            int tt = i >> 4, n = i & 15;
            int row = b * LSEQ + p0 + tt;
            sB[i] = xdbl[(size_t)row * 64 + DTR + n];
            sC[i] = xdbl[(size_t)row * 64 + DTR + DSTATE + n];
        }
        __syncthreads();

        for (int tt = 0; tt < 16; ++tt) {
            const int p   = p0 + tt;
            const int row = b * LSEQ + p;
            const float dtv = dt[(size_t)row * DI + d];
            const float uv  = u[(size_t)row * DI + d];
            float y = 0.0f;
#pragma unroll
            for (int n = 0; n < 16; ++n) {
                float dA = __expf(dtv * Ac[n]);
                h[n] = dA * h[n] + (dtv * sB[tt * 16 + n]) * uv;
                y   += h[n] * sC[tt * 16 + n];
            }
            y += uv * Dv;

            const int l = map_pos(branch, p);
            const float z = xz[(size_t)(b * LSEQ + l) * (2 * DI) + DI + d];
            y *= siluf(z);

            size_t oi = (size_t)(b * LSEQ + l) * DI + d;
            if (accumulate) ysum[oi] += y; else ysum[oi] = y;
        }
    }
}

// -----------------------------------------------------------------------
// Row LayerNorm over DM=512, one block (256 threads, 2 elems each) per row
// -----------------------------------------------------------------------
__global__ __launch_bounds__(256) void layernorm_kernel(
    const float* __restrict__ X, const float* __restrict__ gamma,
    const float* __restrict__ beta, float* __restrict__ out)
{
    __shared__ float r1[256];
    __shared__ float r2[256];
    const int row = blockIdx.x;
    const float* xr = X + (size_t)row * DM;
    float a = xr[threadIdx.x];
    float c = xr[threadIdx.x + 256];
    r1[threadIdx.x] = a + c;
    r2[threadIdx.x] = a * a + c * c;
    __syncthreads();
    for (int off = 128; off > 0; off >>= 1) {
        if (threadIdx.x < off) {
            r1[threadIdx.x] += r1[threadIdx.x + off];
            r2[threadIdx.x] += r2[threadIdx.x + off];
        }
        __syncthreads();
    }
    const float mu  = r1[0] * (1.0f / DM);
    const float var = r2[0] * (1.0f / DM) - mu * mu;
    const float rs  = rsqrtf(var + 1e-5f);
    out[(size_t)row * DM + threadIdx.x]       = (a - mu) * rs * gamma[threadIdx.x]       + beta[threadIdx.x];
    out[(size_t)row * DM + threadIdx.x + 256] = (c - mu) * rs * gamma[threadIdx.x + 256] + beta[threadIdx.x + 256];
}

// -----------------------------------------------------------------------
extern "C" void kernel_launch(void* const* d_in, const int* in_sizes, int n_in,
                              void* d_out, int out_size, void* d_ws, size_t ws_size,
                              hipStream_t stream) {
    (void)in_sizes; (void)n_in; (void)out_size; (void)ws_size;

    const float* x        = (const float*)d_in[0];   // (B,L,DM)
    const float* in_proj  = (const float*)d_in[1];   // (2*DI, DM)
    const float* x_projw  = (const float*)d_in[8];   // (DTR+2*DS, DI) = (64,1024)
    const float* dt_projw = (const float*)d_in[9];   // (DI, DTR)
    const float* dt_projb = (const float*)d_in[10];  // (DI)
    const float* out_projw= (const float*)d_in[17];  // (DM, DI)
    const float* ln_gamma = (const float*)d_in[18];
    const float* ln_beta  = (const float*)d_in[19];

    // workspace layout (floats)
    float* ws     = (float*)d_ws;
    float* xz     = ws;                               // TOK x 2*DI (token-major)
    float* u      = xz     + (size_t)TOK * 2 * DI;    // TOK x DI (per-branch, reused)
    float* xdbl   = u      + (size_t)TOK * DI;        // TOK x 64
    float* dtb    = xdbl   + (size_t)TOK * 64;        // TOK x DI
    float* ysum   = dtb    + (size_t)TOK * DI;        // TOK x DI (original order)
    float* outpre = ysum   + (size_t)TOK * DI;        // TOK x DM

    dim3 blk(256);

    // in_proj: xz = x @ in_proj^T   (2048x512 @ 512x2048)
    gemm_wmma_bf16<<<dim3(TOK / 128, (2 * DI) / 64), blk, 0, stream>>>(
        x, DM, in_proj, DM, xz, 2 * DI, DM, nullptr, 0);

    for (int br = 0; br < 3; ++br) {
        const float* cw = (const float*)d_in[2 + 2 * br];   // conv weights
        const float* cb = (const float*)d_in[3 + 2 * br];   // conv bias
        const float* Al = (const float*)d_in[11 + br];      // A_log variant
        const float* Dv = (const float*)d_in[14 + br];      // D variant

        conv_silu_kernel<<<(TOK * DI) / 256, blk, 0, stream>>>(xz, cw, cb, u, br);

        // x_dbl = u @ x_proj^T   (2048x1024 @ 1024x64)
        gemm_wmma_bf16<<<dim3(TOK / 128, 64 / 64), blk, 0, stream>>>(
            u, DI, x_projw, DI, xdbl, 64, DI, nullptr, 0);

        // dt = softplus(x_dbl[:, :32] @ dt_proj^T + b)   (K=32, N=1024)
        gemm_wmma_bf16<<<dim3(TOK / 128, DI / 64), blk, 0, stream>>>(
            xdbl, 64, dt_projw, DTR, dtb, DI, DTR, dt_projb, 1);

        // selective scan + gate + scatter to original positions
        scan_kernel<<<BATCH * (DI / 256), blk, 0, stream>>>(
            dtb, xdbl, u, xz, Al, Dv, ysum, br, br != 0);
    }

    // out = ysum @ out_proj^T   (2048x1024 @ 1024x512)
    gemm_wmma_bf16<<<dim3(TOK / 128, DM / 64), blk, 0, stream>>>(
        ysum, DI, out_projw, DI, outpre, DM, DI, nullptr, 0);

    layernorm_kernel<<<TOK, blk, 0, stream>>>(outpre, ln_gamma, ln_beta, (float*)d_out);
}
